// Decoder8to4_30348238913927
// MI455X (gfx1250) — matrix operand order repaired and checked by reading, not compiled, not run on hardware
//
#include <hip/hip_runtime.h>
#include <hip/hip_bf16.h>

#define BB 4096
#define HH 1024
#define ZZ 256
#define OO 128
#define GG (3 * HH)
#define TT 32

typedef __bf16 bf16_t;
typedef __attribute__((ext_vector_type(16))) __bf16 v16bf;
typedef __attribute__((ext_vector_type(8)))  __bf16 v8bf;
typedef __attribute__((ext_vector_type(8)))  float   v8f;

// ---------------------------------------------------------------- helpers

static __device__ __forceinline__ bf16_t f2bf(float f) {
    union { float f; unsigned u; } v; v.f = f;
    unsigned r = v.u + 0x7FFFu + ((v.u >> 16) & 1u);   // round-to-nearest-even
    unsigned short h = (unsigned short)(r >> 16);
    return __builtin_bit_cast(__bf16, h);
}

static __device__ __forceinline__ float sigmoidf_(float x) {
    return 1.0f / (1.0f + __expf(-x));
}

// A fragment: 16x32 (MxK) bf16, row-major source A[m, k] with leading dim lda.
// Layout: lane L (M = L&15) holds K = kb..kb+15,
// where kb = k0 + (L>>4)*16.
static __device__ __forceinline__ v16bf load_afrag(const bf16_t* A, int lda,
                                                   int m0, int k0) {
    int lane = threadIdx.x & 31;
    int m  = m0 + (lane & 15);
    int kb = k0 + ((lane >> 4) << 4);
    const bf16_t* p = A + (size_t)m * lda + kb;
    return *(const v16bf*)(p);
}

// B fragment: 32x16 (KxN) bf16 taken from weight W[N, K] row-major (W @ x.T
// convention => column n of B is row n of W). Lane L holds N = n0+(L&15),
// K = k0 + (L>>4)*16 + c for c=0..15 -> one contiguous 32B load.
static __device__ __forceinline__ v16bf load_bfrag(const bf16_t* W, int ldb,
                                                   int n0, int k0) {
    int lane = threadIdx.x & 31;
    int n  = n0 + (lane & 15);
    int kb = k0 + ((lane >> 4) << 4);
    return *(const v16bf*)(W + (size_t)n * ldb + kb);
}

static __device__ __forceinline__ v8f wmma_bf16(v16bf a, v16bf b, v8f c) {
    return __builtin_amdgcn_wmma_f32_16x16x32_bf16(
        /*neg_a=*/false, a, /*neg_b=*/false, b,
        /*c_mod=*/(short)0, c, /*reuse_a=*/false, /*reuse_b=*/false);
}

// ---------------------------------------------------------------- kernels

__global__ void cvt_f32_bf16_kernel(const float* __restrict__ src,
                                    bf16_t* __restrict__ dst, int n) {
    int i = blockIdx.x * blockDim.x + threadIdx.x;
    if (i < n) dst[i] = f2bf(src[i]);
}

// Wih [3H, 128+256] -> wout [3H,128] bf16, wz [3H,256] bf16
__global__ void split_wih_kernel(const float* __restrict__ wih,
                                 bf16_t* __restrict__ wout,
                                 bf16_t* __restrict__ wz) {
    int i = blockIdx.x * blockDim.x + threadIdx.x;
    int total = GG * (OO + ZZ);
    if (i >= total) return;
    int r = i / (OO + ZZ);
    int c = i - r * (OO + ZZ);
    bf16_t v = f2bf(wih[i]);
    if (c < OO) wout[(size_t)r * OO + c] = v;
    else        wz[(size_t)r * ZZ + (c - OO)] = v;
}

// SOS token: out0 = 0 except last column = 1
__global__ void init_out0_kernel(bf16_t* __restrict__ outbf) {
    int i = blockIdx.x * blockDim.x + threadIdx.x;
    if (i >= BB * OO) return;
    int c = i & (OO - 1);
    outbf[i] = f2bf(c == OO - 1 ? 1.0f : 0.0f);
}

// Generic C = act(A[M,K] @ W[N,K]^T + bias), register-blocked:
// each wave computes a (16*MB) x (16*NB) tile of C.
// act: 0 = none, 1 = tanh. Cf (fp32, ld=N) and/or Cb (bf16, ld=N) optional.
template <int MB, int NB>
__global__ void gemm_bias_act_kernel(const bf16_t* __restrict__ A, int lda,
                                     const bf16_t* __restrict__ W, int ldb,
                                     const float* __restrict__ bias,
                                     float* __restrict__ Cf,
                                     bf16_t* __restrict__ Cb,
                                     int M, int N, int K, int act) {
    int wave = (blockIdx.x * blockDim.x + threadIdx.x) >> 5;
    int ntn  = N / (16 * NB);
    int mt   = wave / ntn;
    int nt   = wave - mt * ntn;
    if (mt >= M / (16 * MB)) return;
    int m0 = mt * 16 * MB, n0 = nt * 16 * NB;

    v8f acc[MB][NB];
#pragma unroll
    for (int mi = 0; mi < MB; ++mi)
#pragma unroll
        for (int ni = 0; ni < NB; ++ni) acc[mi][ni] = (v8f){};

    for (int k = 0; k < K; k += 32) {
        v16bf a[MB], b[NB];
#pragma unroll
        for (int mi = 0; mi < MB; ++mi) a[mi] = load_afrag(A, lda, m0 + 16 * mi, k);
#pragma unroll
        for (int ni = 0; ni < NB; ++ni) b[ni] = load_bfrag(W, ldb, n0 + 16 * ni, k);
#pragma unroll
        for (int mi = 0; mi < MB; ++mi)
#pragma unroll
            for (int ni = 0; ni < NB; ++ni)
                acc[mi][ni] = wmma_bf16(a[mi], b[ni], acc[mi][ni]);
    }

    int lane = threadIdx.x & 31;
#pragma unroll
    for (int mi = 0; mi < MB; ++mi) {
#pragma unroll
        for (int ni = 0; ni < NB; ++ni) {
            int col = n0 + 16 * ni + (lane & 15);
            float bv = bias ? bias[col] : 0.0f;
#pragma unroll
            for (int v = 0; v < 8; ++v) {
                int row = m0 + 16 * mi + v + ((lane >> 4) << 3);
                float x = acc[mi][ni][v] + bv;
                if (act == 1) x = tanhf(x);
                size_t idx = (size_t)row * N + col;
                if (Cf) Cf[idx] = x;
                if (Cb) Cb[idx] = f2bf(x);
            }
        }
    }
}

// Fused GRU gate step, register-blocked 32(M) x 16(N of H-space) per wave.
// r,z gates chain both GEMMs into one accumulator (only the sum is needed);
// the n-gate keeps input (i_n) and hidden (h_n) parts separate per GRU math.
// gicon = z @ Wih[:,128:].T + bih (precomputed; covers gi's z part + bias).
__global__ void gru_gate_kernel(const bf16_t* __restrict__ hbf_in,   // [B,H]
                                const bf16_t* __restrict__ outbf,    // [B,O]
                                const bf16_t* __restrict__ whh,      // [3H,H]
                                const bf16_t* __restrict__ wih_out,  // [3H,O]
                                const float*  __restrict__ gicon,    // [B,3H]
                                const float*  __restrict__ bhh,      // [3H]
                                float*        __restrict__ h,        // [B,H]
                                bf16_t*       __restrict__ hbf_out)  // [B,H]
{
    constexpr int MB = 2;
    int wave = (blockIdx.x * blockDim.x + threadIdx.x) >> 5;
    const int ntn = HH >> 4;                // 64 N-tiles across H
    int mt = wave / ntn;
    int nt = wave - mt * ntn;
    int m0 = mt * 16 * MB, n0 = nt << 4;

    v8f ar[MB], az[MB], ain[MB], ahn[MB];
#pragma unroll
    for (int mi = 0; mi < MB; ++mi) {
        ar[mi] = (v8f){}; az[mi] = (v8f){}; ain[mi] = (v8f){}; ahn[mi] = (v8f){};
    }

    // hidden path: K = 1024; B-frags (3 gate columns) shared by MB A-frags
    for (int k = 0; k < HH; k += 32) {
        v16bf br = load_bfrag(whh, HH, n0,          k);
        v16bf bz = load_bfrag(whh, HH, n0 + HH,     k);
        v16bf bn = load_bfrag(whh, HH, n0 + 2 * HH, k);
#pragma unroll
        for (int mi = 0; mi < MB; ++mi) {
            v16bf a = load_afrag(hbf_in, HH, m0 + 16 * mi, k);
            ar[mi]  = wmma_bf16(a, br, ar[mi]);
            az[mi]  = wmma_bf16(a, bz, az[mi]);
            ahn[mi] = wmma_bf16(a, bn, ahn[mi]);
        }
    }
    // input path (prev output only; z part folded into gicon): K = 128
    for (int k = 0; k < OO; k += 32) {
        v16bf br = load_bfrag(wih_out, OO, n0,          k);
        v16bf bz = load_bfrag(wih_out, OO, n0 + HH,     k);
        v16bf bn = load_bfrag(wih_out, OO, n0 + 2 * HH, k);
#pragma unroll
        for (int mi = 0; mi < MB; ++mi) {
            v16bf a = load_afrag(outbf, OO, m0 + 16 * mi, k);
            ar[mi]  = wmma_bf16(a, br, ar[mi]);
            az[mi]  = wmma_bf16(a, bz, az[mi]);
            ain[mi] = wmma_bf16(a, bn, ain[mi]);
        }
    }

    int lane = threadIdx.x & 31;
    int col  = n0 + (lane & 15);
    float brr = bhh[col];
    float bzz = bhh[col + HH];
    float bnn = bhh[col + 2 * HH];
#pragma unroll
    for (int mi = 0; mi < MB; ++mi) {
#pragma unroll
        for (int v = 0; v < 8; ++v) {
            int row = m0 + 16 * mi + v + ((lane >> 4) << 3);
            size_t gib = (size_t)row * GG;
            float gr = gicon[gib + col];
            float gz = gicon[gib + col + HH];
            float gn = gicon[gib + col + 2 * HH];
            float r  = sigmoidf_(ar[mi][v] + gr + brr);
            float zg = sigmoidf_(az[mi][v] + gz + bzz);
            float nn = tanhf(ain[mi][v] + gn + r * (ahn[mi][v] + bnn));
            size_t hidx = (size_t)row * HH + col;
            float hp = h[hidx];
            float hv = (1.0f - zg) * nn + zg * hp;
            h[hidx]       = hv;
            hbf_out[hidx] = f2bf(hv);
        }
    }
}

// o = h @ Wo.T + bo -> d_out[b, t, :] (fp32) and outbf (bf16 next-step x);
// register-blocked 32(M) x 16(N) per wave.
__global__ void out_proj_kernel(const bf16_t* __restrict__ hbf,  // [B,H]
                                const bf16_t* __restrict__ wo,   // [O,H]
                                const float*  __restrict__ bo,   // [O]
                                float*        __restrict__ dout, // [B,T,O]
                                bf16_t*       __restrict__ outbf,// [B,O]
                                int t) {
    constexpr int MB = 2;
    int wave = (blockIdx.x * blockDim.x + threadIdx.x) >> 5;
    const int ntn = OO >> 4;                // 8 N-tiles across O
    int mt = wave / ntn;
    int nt = wave - mt * ntn;
    int m0 = mt * 16 * MB, n0 = nt << 4;

    v8f acc[MB];
#pragma unroll
    for (int mi = 0; mi < MB; ++mi) acc[mi] = (v8f){};

    for (int k = 0; k < HH; k += 32) {
        v16bf b = load_bfrag(wo, HH, n0, k);
#pragma unroll
        for (int mi = 0; mi < MB; ++mi) {
            acc[mi] = wmma_bf16(load_afrag(hbf, HH, m0 + 16 * mi, k), b, acc[mi]);
        }
    }
    int lane = threadIdx.x & 31;
    int col  = n0 + (lane & 15);
    float bv = bo[col];
#pragma unroll
    for (int mi = 0; mi < MB; ++mi) {
#pragma unroll
        for (int v = 0; v < 8; ++v) {
            int row = m0 + 16 * mi + v + ((lane >> 4) << 3);
            float x = acc[mi][v] + bv;
            dout[(size_t)row * (TT * OO) + (size_t)t * OO + col] = x;
            outbf[(size_t)row * OO + col] = f2bf(x);
        }
    }
}

// ---------------------------------------------------------------- launcher

extern "C" void kernel_launch(void* const* d_in, const int* in_sizes, int n_in,
                              void* d_out, int out_size, void* d_ws, size_t ws_size,
                              hipStream_t stream) {
    (void)in_sizes; (void)n_in; (void)out_size; (void)ws_size;

    char* ws = (char*)d_ws;
    size_t off = 0;
    auto alloc = [&](size_t bytes) -> void* {
        void* p = ws + off;
        off = (off + bytes + 255) & ~(size_t)255;
        return p;
    };

    const int ELT = 256;   // elementwise block
    const int GB  = 256;   // gemm block (8 waves)

    for (int d = 0; d < 2; ++d) {
        const float* z    = (const float*)d_in[d == 0 ? 0 : 1];
        const float* Wi   = (const float*)d_in[d == 0 ? 4 : 6];
        const float* bi   = (const float*)d_in[d == 0 ? 5 : 7];
        const float* Wih  = (const float*)d_in[d == 0 ? 8 : 12];
        const float* Whh  = (const float*)d_in[d == 0 ? 9 : 13];
        const float* bih  = (const float*)d_in[d == 0 ? 10 : 14];
        const float* bhh  = (const float*)d_in[d == 0 ? 11 : 15];
        const float* Wo   = (const float*)d_in[d == 0 ? 16 : 18];
        const float* bo   = (const float*)d_in[d == 0 ? 17 : 19];

        bf16_t* wi_bf   = (bf16_t*)alloc((size_t)HH * ZZ * 2);
        bf16_t* wihO_bf = (bf16_t*)alloc((size_t)GG * OO * 2);
        bf16_t* wihZ_bf = (bf16_t*)alloc((size_t)GG * ZZ * 2);
        bf16_t* whh_bf  = (bf16_t*)alloc((size_t)GG * HH * 2);
        bf16_t* wo_bf   = (bf16_t*)alloc((size_t)OO * HH * 2);
        bf16_t* z_bf    = (bf16_t*)alloc((size_t)BB * ZZ * 2);
        bf16_t* hbf[2];
        hbf[0]          = (bf16_t*)alloc((size_t)BB * HH * 2);
        hbf[1]          = (bf16_t*)alloc((size_t)BB * HH * 2);
        bf16_t* out_bf  = (bf16_t*)alloc((size_t)BB * OO * 2);
        float*  hf      = (float*)alloc((size_t)BB * HH * 4);
        float*  gicon   = (float*)alloc((size_t)BB * GG * 4);

        float* dout = (float*)d_out + (size_t)d * BB * TT * OO;

        // --- weight / latent conversion (fp32 -> bf16) ---
        cvt_f32_bf16_kernel<<<(HH * ZZ + ELT - 1) / ELT, ELT, 0, stream>>>(
            Wi, wi_bf, HH * ZZ);
        split_wih_kernel<<<(GG * (OO + ZZ) + ELT - 1) / ELT, ELT, 0, stream>>>(
            Wih, wihO_bf, wihZ_bf);
        cvt_f32_bf16_kernel<<<(GG * HH + ELT - 1) / ELT, ELT, 0, stream>>>(
            Whh, whh_bf, GG * HH);
        cvt_f32_bf16_kernel<<<(OO * HH + ELT - 1) / ELT, ELT, 0, stream>>>(
            Wo, wo_bf, OO * HH);
        cvt_f32_bf16_kernel<<<(BB * ZZ + ELT - 1) / ELT, ELT, 0, stream>>>(
            z, z_bf, BB * ZZ);
        init_out0_kernel<<<(BB * OO + ELT - 1) / ELT, ELT, 0, stream>>>(out_bf);

        // --- h0 = tanh(z @ Wi.T + bi)   [4096,1024], K=256, 32x32 tiles ---
        {
            int waves = (BB / 32) * (HH / 32);            // 4096
            gemm_bias_act_kernel<2, 2><<<waves / 8, GB, 0, stream>>>(
                z_bf, ZZ, wi_bf, ZZ, bi, hf, hbf[0], BB, HH, ZZ, /*tanh*/1);
        }
        // --- gicon = z @ Wih[:,128:].T + bih   [4096,3072], K=256 ---
        {
            int waves = (BB / 32) * (GG / 32);            // 12288
            gemm_bias_act_kernel<2, 2><<<waves / 8, GB, 0, stream>>>(
                z_bf, ZZ, wihZ_bf, ZZ, bih, gicon, (bf16_t*)nullptr,
                BB, GG, ZZ, /*none*/0);
        }

        // --- 32 recurrent steps (stream order enforces the dependence) ---
        int gate_waves = (BB / 32) * (HH / 16);   // 8192
        int out_waves  = (BB / 32) * (OO / 16);   // 1024
        for (int t = 0; t < TT; ++t) {
            gru_gate_kernel<<<gate_waves / 8, GB, 0, stream>>>(
                hbf[t & 1], out_bf, whh_bf, wihO_bf, gicon, bhh,
                hf, hbf[(t + 1) & 1]);
            out_proj_kernel<<<out_waves / 8, GB, 0, stream>>>(
                hbf[(t + 1) & 1], wo_bf, bo, dout, out_bf, t);
        }
    }
}
